// TabTransformerGNN_83339545411982
// MI455X (gfx1250) — matrix operand order repaired and checked by reading, not compiled, not run on hardware
//
#include <hip/hip_runtime.h>
#include <hip/hip_bf16.h>

// ---------------------------------------------------------------------------
// TabTransformer + hetero-SAGE for MI455X (gfx1250, wave32, WMMA).
// All dense GEMMs ([*,128]x[128,128] and the 643-K decoder) run through a
// tiled v_wmma_f32_16x16x32_f16 kernel: f16 operands, f32 accumulation.
// Weights are pre-transposed/converted to f16 once per GEMM (L2-resident),
// so the hot loop is pure b128 vmem -> b128 ds -> wmma.
// ---------------------------------------------------------------------------

typedef __attribute__((ext_vector_type(16))) _Float16 v16h;
typedef __attribute__((ext_vector_type(8)))  _Float16 v8h;
typedef __attribute__((ext_vector_type(8)))  float    v8f;

#define NCATC 5
#define DC    128
#define MT    64          // GEMM M tile per workgroup
#define NA_C  50000
#define NB_C  50000
#define NC_C  100000
#define E_C   500000

// ---------------------------------------------------------------------------
// Weight prep: Wt[n][k] = (f16) W[k][n], zero-padded to Kpad (multiple of 32).
// Tiny (<=128x672); reads coalesced along n.
// ---------------------------------------------------------------------------
__global__ __launch_bounds__(256)
void transpose_w16_kernel(const float* __restrict__ W, _Float16* __restrict__ Wt,
                          int K, int Kpad)
{
    int idx = blockIdx.x * 256 + threadIdx.x;
    if (idx >= 128 * Kpad) return;
    int n = idx & 127;
    int k = idx >> 7;
    Wt[(size_t)n * Kpad + k] = (k < K) ? (_Float16)W[(size_t)k * 128 + n]
                                       : (_Float16)0.f;
}

// ---------------------------------------------------------------------------
// WMMA GEMM: C[M,128] = A[M,Kpad](f16, lda, zero-padded) @ Wt[128,Kpad](f16)
// (+bias)(+resid)(relu)(f16 mirror). 8 waves; wave w owns N cols [w*16,w*16+16)
// and 4 M sub-tiles of 16. K consumed in chunks of 32 staged through LDS.
// ---------------------------------------------------------------------------
template<bool BIAS, bool RELU, bool RESID, bool W16OUT>
__global__ __launch_bounds__(256)
void gemm128_wmma(const _Float16* __restrict__ A, int lda,
                  const _Float16* __restrict__ Wt, int Kpad,
                  const float* __restrict__ bias,
                  const float* resid,          // may alias C
                  float* C,
                  _Float16* __restrict__ C16,
                  int M)
{
    __shared__ _Float16 sA[MT][32];        // A chunk, row-major
    __shared__ _Float16 sWt[128][40];      // Wt chunk [n][k], padded rows

    const int tid  = threadIdx.x;
    const int wave = tid >> 5;
    const int lane = tid & 31;
    const int m0   = blockIdx.x * MT;
    const int n0   = wave * 16;

    // per-thread staging coordinates (loop-invariant)
    const int ar   = tid >> 2;             // A row in tile (0..63)
    const int ac   = (tid & 3) << 3;       // A col group (0,8,16,24)
    const int arow = m0 + ar;
    const int wn   = tid >> 1;             // Wt row n (0..127)
    const int wc   = (tid & 1) << 4;       // Wt col group (0,16)

    v8f acc[4] = {};

    const int kTiles = Kpad >> 5;
    for (int kt = 0; kt < kTiles; ++kt) {
        const int kc = kt << 5;
        if (kt + 1 < kTiles)   // pull next Wt chunk toward L0 early
            __builtin_prefetch(Wt + (size_t)wn * Kpad + kc + 32, 0, 1);
        __syncthreads();
        // --- stage A chunk: one predicated b128 per thread ---
        {
            v8h av = {};
            if (arow < M)
                av = *(const v8h*)(A + (size_t)arow * lda + kc + ac);
            *(v8h*)&sA[ar][ac] = av;
        }
        // --- stage Wt chunk: two b128 per thread, no guards ---
        {
            const _Float16* ws = Wt + (size_t)wn * Kpad + kc + wc;
            *(v8h*)&sWt[wn][wc]     = *(const v8h*)(ws);
            *(v8h*)&sWt[wn][wc + 8] = *(const v8h*)(ws + 8);
        }
        __syncthreads();

        // B fragment (32x16): lane holds col n0+(lane&15); K groups
        // {kb..kb+7, kb+16..kb+23} with kb = lane<16 ? 0 : 8.
        v16h bfrag;
        {
            int n  = n0 + (lane & 15);
            int kb = (lane < 16) ? 0 : 8;
#pragma unroll
            for (int i = 0; i < 16; ++i) bfrag[i] = sWt[n][kb + (i < 8 ? i : i + 8)];
        }
#pragma unroll
        for (int ms = 0; ms < 4; ++ms) {
            // A fragment (16x32): lane row = (lane&15); K groups
            // {kb..kb+7, kb+16..kb+23} with kb = lane<16 ? 0 : 8.
            v16h afrag;
            int r  = (ms << 4) + (lane & 15);
            int kb = (lane < 16) ? 0 : 8;
#pragma unroll
            for (int i = 0; i < 8; ++i) afrag[i]     = sA[r][kb + i];
#pragma unroll
            for (int i = 0; i < 8; ++i) afrag[8 + i] = sA[r][kb + 16 + i];
            acc[ms] = __builtin_amdgcn_wmma_f32_16x16x32_f16(
                          false, afrag, false, bfrag,
                          (short)0, acc[ms], false, false);
        }
    }

    // Epilogue. C/D layout: VGPR r, lanes<16 -> M=r, lanes>=16 -> M=r+8.
    const int nOut = n0 + (lane & 15);
#pragma unroll
    for (int ms = 0; ms < 4; ++ms) {
#pragma unroll
        for (int r = 0; r < 8; ++r) {
            int row = m0 + (ms << 4) + r + ((lane < 16) ? 0 : 8);
            if (row < M) {
                float v = acc[ms][r];
                if (BIAS)  v += bias[nOut];
                if (RESID) v += resid[(size_t)row * 128 + nOut];
                if (RELU)  v = v > 0.f ? v : 0.f;
                C[(size_t)row * 128 + nOut] = v;
                if (W16OUT) C16[(size_t)row * 128 + nOut] = (_Float16)v;
            }
        }
    }
}

// ---------------------------------------------------------------------------
// LayerNorm over 128-wide rows, f32 in -> f16 out. One wave per row.
// ---------------------------------------------------------------------------
__global__ __launch_bounds__(256)
void layernorm128_kernel(const float* __restrict__ x,
                         const float* __restrict__ g,
                         const float* __restrict__ b,
                         _Float16* __restrict__ z, int M)
{
    int row  = blockIdx.x * 8 + (threadIdx.x >> 5);
    int lane = threadIdx.x & 31;
    if (row >= M) return;
    const float* xr = x + (size_t)row * 128;
    float4 v = ((const float4*)xr)[lane];
    float s  = v.x + v.y + v.z + v.w;
    float s2 = v.x * v.x + v.y * v.y + v.z * v.z + v.w * v.w;
#pragma unroll
    for (int m = 16; m; m >>= 1) {
        s  += __shfl_xor(s,  m, 32);
        s2 += __shfl_xor(s2, m, 32);
    }
    float mean = s * (1.f / 128.f);
    float var  = s2 * (1.f / 128.f) - mean * mean;
    float rstd = rsqrtf(var + 1e-5f);
    int c = lane * 4;
    z[(size_t)row * 128 + c + 0] = (_Float16)((v.x - mean) * rstd * g[c + 0] + b[c + 0]);
    z[(size_t)row * 128 + c + 1] = (_Float16)((v.y - mean) * rstd * g[c + 1] + b[c + 1]);
    z[(size_t)row * 128 + c + 2] = (_Float16)((v.z - mean) * rstd * g[c + 2] + b[c + 2]);
    z[(size_t)row * 128 + c + 3] = (_Float16)((v.w - mean) * rstd * g[c + 3] + b[c + 3]);
}

// ---------------------------------------------------------------------------
// Embedding gather: h[n][c][0:126] = emb[c][x_cat[n][c]], h[n][c][126:128]=pad
// ---------------------------------------------------------------------------
__global__ __launch_bounds__(128)
void embed_kernel(const int* __restrict__ x_cat,
                  const float* __restrict__ emb,   // [5,100,126]
                  const float* __restrict__ pad,   // [5,2]
                  float* __restrict__ h, int N)
{
    int nc = blockIdx.x;                 // n*5 + c
    int c  = nc % NCATC;
    int n  = nc / NCATC;
    int d  = threadIdx.x;
    float v;
    if (d < 126) v = emb[((size_t)c * 100 + x_cat[n * NCATC + c]) * 126 + d];
    else         v = pad[c * 2 + (d - 126)];
    h[(size_t)nc * 128 + d] = v;
}

// ---------------------------------------------------------------------------
// Attention over 5 tokens x 4 heads x 32 dims. One thread per (node, head).
// q/k/v rows are [5,128] with head h at cols [h*32, h*32+32).
// ---------------------------------------------------------------------------
__global__ __launch_bounds__(256)
void attn_kernel(const float* __restrict__ q, const float* __restrict__ k,
                 const float* __restrict__ v, _Float16* __restrict__ o, int N)
{
    int idx = blockIdx.x * blockDim.x + threadIdx.x;
    if (idx >= N * 4) return;
    int n  = idx >> 2, hd = idx & 3;
    const float scale = 0.17677669529663687f;   // 1/sqrt(32)
    const float* qb = q + (size_t)n * 640 + hd * 32;
    const float* kb = k + (size_t)n * 640 + hd * 32;
    const float* vb = v + (size_t)n * 640 + hd * 32;
    for (int s = 0; s < 5; ++s) {
        float sc[5], mx = -1e30f;
#pragma unroll
        for (int t = 0; t < 5; ++t) {
            float a = 0.f;
            for (int d = 0; d < 32; ++d) a += qb[s * 128 + d] * kb[t * 128 + d];
            sc[t] = a * scale;
            mx = fmaxf(mx, sc[t]);
        }
        float sum = 0.f;
#pragma unroll
        for (int t = 0; t < 5; ++t) { sc[t] = __expf(sc[t] - mx); sum += sc[t]; }
        float inv = 1.f / sum;
        for (int d = 0; d < 32; ++d) {
            float a = 0.f;
#pragma unroll
            for (int t = 0; t < 5; ++t) a += sc[t] * vb[t * 128 + d];
            o[(size_t)n * 640 + s * 128 + hd * 32 + d] = (_Float16)(a * inv);
        }
    }
}

// ---------------------------------------------------------------------------
// feat[n] = concat(f16(h[n].flatten()), LN3(x_num[n]), zeros) -> width 672
// ---------------------------------------------------------------------------
__global__ __launch_bounds__(256)
void feat_kernel(const float* __restrict__ h, const float* __restrict__ xnum,
                 const float* __restrict__ g3, const float* __restrict__ b3,
                 _Float16* __restrict__ feat, int N)
{
    int n = blockIdx.x;
    for (int i = threadIdx.x; i < 672; i += blockDim.x) {
        _Float16 val;
        if (i < 640) {
            val = (_Float16)h[(size_t)n * 640 + i];
        } else if (i < 643) {
            float a0 = xnum[n * 3 + 0], a1 = xnum[n * 3 + 1], a2 = xnum[n * 3 + 2];
            float m  = (a0 + a1 + a2) * (1.f / 3.f);
            float vv = ((a0 - m) * (a0 - m) + (a1 - m) * (a1 - m) +
                        (a2 - m) * (a2 - m)) * (1.f / 3.f);
            float r  = rsqrtf(vv + 1e-5f);
            int   j  = i - 640;
            float aj = (j == 0) ? a0 : (j == 1) ? a1 : a2;
            val = (_Float16)((aj - m) * r * g3[j] + b3[j]);
        } else {
            val = (_Float16)0.f;
        }
        feat[(size_t)n * 672 + i] = val;
    }
}

// ---------------------------------------------------------------------------
// SAGE helpers
// ---------------------------------------------------------------------------
__global__ __launch_bounds__(256)
void zero_f32_kernel(float* __restrict__ p, int n)
{
    int i = blockIdx.x * 256 + threadIdx.x;
    if (i < n) p[i] = 0.f;
}

__global__ __launch_bounds__(256)
void scatter_add_kernel(const float* __restrict__ xsrc,
                        const int* __restrict__ src, const int* __restrict__ dst,
                        float* agg, float* cnt, int E)
{
    int e = blockIdx.x * 2 + (threadIdx.x >> 7);
    int d = threadIdx.x & 127;
    if (e >= E) return;
    int s = src[e], t = dst[e];
    atomicAdd(&agg[(size_t)t * 128 + d], xsrc[(size_t)s * 128 + d]);
    if (d == 0) atomicAdd(&cnt[t], 1.f);
}

__global__ __launch_bounds__(256)
void mean16_kernel(const float* __restrict__ agg, const float* __restrict__ cnt,
                   _Float16* __restrict__ m16, int N)
{
    int i = blockIdx.x * 256 + threadIdx.x;
    if (i >= N * 128) return;
    float c = fmaxf(cnt[i >> 7], 1.f);
    m16[i] = (_Float16)(agg[i] / c);
}

__global__ __launch_bounds__(256)
void cvt_f16_kernel(const float* __restrict__ s, _Float16* __restrict__ d, int n)
{
    int i = blockIdx.x * 256 + threadIdx.x;
    if (i < n) d[i] = (_Float16)s[i];
}

// ---------------------------------------------------------------------------
// Head: logits = hA @ Wout[128,16] + bout; softmax over 16. 16 lanes / node.
// ---------------------------------------------------------------------------
__global__ __launch_bounds__(256)
void head_softmax_kernel(const float* __restrict__ hA,
                         const float* __restrict__ Wout,
                         const float* __restrict__ bout,
                         float* __restrict__ out, int N)
{
    int lane = threadIdx.x & 31;
    int wv   = (blockIdx.x * blockDim.x + threadIdx.x) >> 5;
    int n    = wv * 2 + (lane >> 4);
    int col  = lane & 15;
    if (n >= N) return;
    float acc = bout[col];
    const float* hr = hA + (size_t)n * 128;
    for (int k = 0; k < 128; ++k) acc += hr[k] * Wout[k * 16 + col];
    float mx = acc;
#pragma unroll
    for (int m = 1; m < 16; m <<= 1) mx = fmaxf(mx, __shfl_xor(mx, m, 32));
    float ex = __expf(acc - mx);
    float sum = ex;
#pragma unroll
    for (int m = 1; m < 16; m <<= 1) sum += __shfl_xor(sum, m, 32);
    out[(size_t)n * 16 + col] = ex / sum;
}

// ---------------------------------------------------------------------------
// Host orchestration
// ---------------------------------------------------------------------------
enum { P_EMB = 0, P_PAD, P_WQ, P_WK, P_WV, P_WO, P_W1, P_B1, P_W2, P_B2,
       P_LN1G, P_LN1B, P_LN2G, P_LN2B, P_LNNG, P_LNNB, P_WDEC, P_BDEC };

enum { I_XCAT_A = 0, I_XNUM_A, I_XCAT_B, I_XNUM_B, I_XCAT_C, I_XNUM_C,
       I_EDGE_BA, I_EDGE_CA, I_EDGE_AB,
       I_PA = 9, I_PB = 27, I_PC = 45,
       I_SAGE = 63,   // BA{Wl,Wr,b}, CA{..}, AB{..}
       I_WOUT = 72, I_BOUT = 73 };

template<bool BIAS, bool RELU, bool RESID, bool W16OUT>
static void gemm(hipStream_t st, const _Float16* A, int lda,
                 const float* W, int K, _Float16* wt,
                 const float* bias, const float* resid, float* C,
                 _Float16* C16, int M)
{
    const int Kpad = (K + 31) & ~31;
    transpose_w16_kernel<<<(128 * Kpad + 255) / 256, 256, 0, st>>>(W, wt, K, Kpad);
    gemm128_wmma<BIAS, RELU, RESID, W16OUT>
        <<<(M + MT - 1) / MT, 256, 0, st>>>(A, lda, wt, Kpad, bias, resid, C, C16, M);
}

static void run_table(hipStream_t st, const int* x_cat, const float* x_num,
                      void* const* P, float* x_out, int N,
                      float* h, _Float16* z16, float* qb, float* kb, float* vb,
                      _Float16* o16, _Float16* ffn16, float* ffnf,
                      _Float16* feat16, _Float16* wt)
{
    const int R = N * NCATC;
    embed_kernel<<<R, 128, 0, st>>>(x_cat, (const float*)P[P_EMB],
                                    (const float*)P[P_PAD], h, N);
    for (int l = 0; l < 2; ++l) {
        const float* Wq = (const float*)P[P_WQ] + (size_t)l * 128 * 128;
        const float* Wk = (const float*)P[P_WK] + (size_t)l * 128 * 128;
        const float* Wv = (const float*)P[P_WV] + (size_t)l * 128 * 128;
        const float* Wo = (const float*)P[P_WO] + (size_t)l * 128 * 128;
        const float* W1 = (const float*)P[P_W1] + (size_t)l * 128 * 128;
        const float* W2 = (const float*)P[P_W2] + (size_t)l * 128 * 128;
        const float* b1 = (const float*)P[P_B1] + (size_t)l * 128;
        const float* b2 = (const float*)P[P_B2] + (size_t)l * 128;

        layernorm128_kernel<<<(R + 7) / 8, 256, 0, st>>>(
            h, (const float*)P[P_LN1G] + l * 128,
            (const float*)P[P_LN1B] + l * 128, z16, R);
        gemm<false, false, false, false>(st, z16, 128, Wq, 128, wt, nullptr, nullptr, qb, nullptr, R);
        gemm<false, false, false, false>(st, z16, 128, Wk, 128, wt, nullptr, nullptr, kb, nullptr, R);
        gemm<false, false, false, false>(st, z16, 128, Wv, 128, wt, nullptr, nullptr, vb, nullptr, R);
        attn_kernel<<<(N * 4 + 255) / 256, 256, 0, st>>>(qb, kb, vb, o16, N);
        gemm<false, false, true, false>(st, o16, 128, Wo, 128, wt, nullptr, h, h, nullptr, R);

        layernorm128_kernel<<<(R + 7) / 8, 256, 0, st>>>(
            h, (const float*)P[P_LN2G] + l * 128,
            (const float*)P[P_LN2B] + l * 128, z16, R);
        gemm<true, true, false, true>(st, z16, 128, W1, 128, wt, b1, nullptr, ffnf, ffn16, R);
        gemm<true, false, true, false>(st, ffn16, 128, W2, 128, wt, b2, h, h, nullptr, R);
    }
    feat_kernel<<<N, 256, 0, st>>>(h, x_num, (const float*)P[P_LNNG],
                                   (const float*)P[P_LNNB], feat16, N);
    gemm<true, false, false, false>(st, feat16, 672, (const float*)P[P_WDEC],
                                    NCATC * DC + 3, wt, (const float*)P[P_BDEC],
                                    nullptr, x_out, nullptr, N);
}

static inline char* bump(char*& p, size_t bytes)
{
    char* r = p;
    p += (bytes + 255) & ~(size_t)255;
    return r;
}

extern "C" void kernel_launch(void* const* d_in, const int* in_sizes, int n_in,
                              void* d_out, int out_size, void* d_ws, size_t ws_size,
                              hipStream_t stream)
{
    (void)in_sizes; (void)n_in; (void)out_size;
    const int NA = NA_C, NB = NB_C, NC = NC_C, E = E_C;
    const int RC = NC * NCATC;       // largest row count (500k)

    char* p = (char*)d_ws;
    float*     xA     = (float*)bump(p, (size_t)NA * 128 * 4);
    float*     xB     = (float*)bump(p, (size_t)NB * 128 * 4);
    float*     xC     = (float*)bump(p, (size_t)NC * 128 * 4);
    float*     hA     = (float*)bump(p, (size_t)NA * 128 * 4);
    float*     agg    = (float*)bump(p, (size_t)NA * 128 * 4);
    float*     cnt    = (float*)bump(p, (size_t)NA * 4);
    _Float16*  mean16 = (_Float16*)bump(p, (size_t)NA * 128 * 2);
    _Float16*  cvt16  = (_Float16*)bump(p, (size_t)NA * 128 * 2);
    _Float16*  wt     = (_Float16*)bump(p, (size_t)128 * 672 * 2);
    // transformer scratch (sized for table C, reused across tables)
    float*     h      = (float*)bump(p, (size_t)RC * 128 * 4);
    _Float16*  z16    = (_Float16*)bump(p, (size_t)RC * 128 * 2);
    float*     qb     = (float*)bump(p, (size_t)RC * 128 * 4);
    float*     kb     = (float*)bump(p, (size_t)RC * 128 * 4);
    float*     vb     = (float*)bump(p, (size_t)RC * 128 * 4);
    _Float16*  o16    = (_Float16*)bump(p, (size_t)RC * 128 * 2);
    _Float16*  ffn16  = (_Float16*)bump(p, (size_t)RC * 128 * 2);
    float*     ffnf   = (float*)bump(p, (size_t)RC * 128 * 4);
    _Float16*  feat16 = (_Float16*)bump(p, (size_t)NC * 672 * 2);
    if ((size_t)(p - (char*)d_ws) > ws_size) return;   // scratch too small

    // --- three TabTransformers ---
    run_table(stream, (const int*)d_in[I_XCAT_A], (const float*)d_in[I_XNUM_A],
              d_in + I_PA, xA, NA, h, z16, qb, kb, vb, o16, ffn16, ffnf, feat16, wt);
    run_table(stream, (const int*)d_in[I_XCAT_B], (const float*)d_in[I_XNUM_B],
              d_in + I_PB, xB, NB, h, z16, qb, kb, vb, o16, ffn16, ffnf, feat16, wt);
    run_table(stream, (const int*)d_in[I_XCAT_C], (const float*)d_in[I_XNUM_C],
              d_in + I_PC, xC, NC, h, z16, qb, kb, vb, o16, ffn16, ffnf, feat16, wt);

    // --- SAGE: B->A then C->A, summed into hA ---
    const int* eBA = (const int*)d_in[I_EDGE_BA];
    const int* eCA = (const int*)d_in[I_EDGE_CA];
    const float* WlBA = (const float*)d_in[I_SAGE + 0];
    const float* WrBA = (const float*)d_in[I_SAGE + 1];
    const float* bBA  = (const float*)d_in[I_SAGE + 2];
    const float* WlCA = (const float*)d_in[I_SAGE + 3];
    const float* WrCA = (const float*)d_in[I_SAGE + 4];
    const float* bCA  = (const float*)d_in[I_SAGE + 5];

    cvt_f16_kernel<<<(NA * 128 + 255) / 256, 256, 0, stream>>>(xA, cvt16, NA * 128);

    zero_f32_kernel<<<(NA * 128 + 255) / 256, 256, 0, stream>>>(agg, NA * 128);
    zero_f32_kernel<<<(NA + 255) / 256, 256, 0, stream>>>(cnt, NA);
    scatter_add_kernel<<<(E + 1) / 2, 256, 0, stream>>>(xB, eBA, eBA + E, agg, cnt, E);
    mean16_kernel<<<(NA * 128 + 255) / 256, 256, 0, stream>>>(agg, cnt, mean16, NA);
    gemm<true,  false, false, false>(stream, mean16, 128, WlBA, 128, wt, bBA, nullptr, hA, nullptr, NA);
    gemm<false, false, true,  false>(stream, cvt16, 128, WrBA, 128, wt, nullptr, hA, hA, nullptr, NA);

    zero_f32_kernel<<<(NA * 128 + 255) / 256, 256, 0, stream>>>(agg, NA * 128);
    zero_f32_kernel<<<(NA + 255) / 256, 256, 0, stream>>>(cnt, NA);
    scatter_add_kernel<<<(E + 1) / 2, 256, 0, stream>>>(xC, eCA, eCA + E, agg, cnt, E);
    mean16_kernel<<<(NA * 128 + 255) / 256, 256, 0, stream>>>(agg, cnt, mean16, NA);
    gemm<true,  false, true,  false>(stream, mean16, 128, WlCA, 128, wt, bCA, hA, hA, nullptr, NA);
    gemm<false, false, true,  false>(stream, cvt16, 128, WrCA, 128, wt, nullptr, hA, hA, nullptr, NA);

    // --- head + softmax ---
    head_softmax_kernel<<<(NA + 15) / 16, 256, 0, stream>>>(
        hA, (const float*)d_in[I_WOUT], (const float*)d_in[I_BOUT],
        (float*)d_out, NA);
}
